// RDHAgent_1546188226868
// MI455X (gfx1250) — matrix operand order repaired and checked by reading, not compiled, not run on hardware
//
#include <hip/hip_runtime.h>
#include <hip/hip_bf16.h>

typedef __attribute__((ext_vector_type(16))) __bf16 v16bf;
typedef __attribute__((ext_vector_type(8)))  __bf16 v8bf;
typedef __attribute__((ext_vector_type(8)))  float  v8f;

#define EPSF 1e-7f

// ---------------------------------------------------------------------------
// Prep: fold softmax(merger) head-average + feat-contraction layout into the
// hypernet second-layer weights.
//   W2rbT[net][ncol = f*64 + o][m]  (bf16, N-major so B fragments are
//                                    contiguous 8-element K runs)
//   b2r  [net][f*64 + o]            (f32)
// W2r[m,f,o] = sum_head mw[head,o] * W2[m, f*256 + head*64 + o], f in [0,10)
// (f == 9 indexes the hypernet-bias slice of the 2560-wide output.)
// ---------------------------------------------------------------------------
__global__ void __launch_bounds__(256)
fold_weights_kernel(const float* __restrict__ merger_w,
                    const float* __restrict__ wA, const float* __restrict__ bA,
                    const float* __restrict__ wE, const float* __restrict__ bE,
                    const float* __restrict__ wL, const float* __restrict__ bL,
                    __bf16* __restrict__ W2rbT, float* __restrict__ b2r)
{
    const int PER_NET = 650 * 64;           // 640*64 weights + 10*64 biases
    int id = blockIdx.x * 256 + threadIdx.x;
    if (id >= 3 * PER_NET) return;
    int net = id / PER_NET;
    int rem = id % PER_NET;

    const float* W2 = (net == 0) ? wA : (net == 1) ? wE : wL;
    const float* B2 = (net == 0) ? bA : (net == 1) ? bE : bL;

    int o = (rem < 640 * 64) ? ((rem / 64) & 63) : ((rem - 640 * 64) & 63);

    // softmax over HEAD=4 of merger_w[head][o]
    float m0 = merger_w[0 * 64 + o], m1 = merger_w[1 * 64 + o];
    float m2 = merger_w[2 * 64 + o], m3 = merger_w[3 * 64 + o];
    float mx = fmaxf(fmaxf(m0, m1), fmaxf(m2, m3));
    float e0 = __expf(m0 - mx), e1 = __expf(m1 - mx);
    float e2 = __expf(m2 - mx), e3 = __expf(m3 - mx);
    float inv = 1.0f / (e0 + e1 + e2 + e3);
    float mw0 = e0 * inv, mw1 = e1 * inv, mw2 = e2 * inv, mw3 = e3 * inv;

    if (rem < 640 * 64) {
        int ncol = rem / 64;
        int m    = rem & 63;
        int f    = ncol / 64;
        const float* row = W2 + (size_t)m * 2560 + f * 256 + o;
        float v = mw0 * row[0] + mw1 * row[64] + mw2 * row[128] + mw3 * row[192];
        W2rbT[(size_t)net * 640 * 64 + (size_t)ncol * 64 + m] = (__bf16)v;
    } else {
        int bidx = rem - 640 * 64;           // f*64 + o
        int f    = bidx / 64;
        const float* row = B2 + f * 256 + o;
        float v = mw0 * row[0] + mw1 * row[64] + mw2 * row[128] + mw3 * row[192];
        b2r[net * 640 + bidx] = v;
    }
}

// row layout inside one slot: 5 M-tiles of 16, each tile one hypernet.
// rows 0..15  : ally  k=0,1  (k*8+j)
// rows 16..39 : enemy k=2..4 ; rows 40..47 pad
// rows 48..71 : lm    k=5..7 ; rows 72..79 pad
__device__ __forceinline__ int rowKJ(int row, int& k, int& j)
{
    if (row < 16) { k = row >> 3;            j = row & 7; return 1; }
    if (row < 40) { int r = row - 16; k = 2 + (r >> 3); j = r & 7; return 1; }
    if (row < 48) return 0;
    if (row < 72) { int r = row - 48; k = 5 + (r >> 3); j = r & 7; return 1; }
    return 0;
}

__device__ __forceinline__ int rowOf(int k, int j)
{
    return (k < 2) ? (k * 8 + j) : (k < 5) ? (16 + (k - 2) * 8 + j)
                                           : (48 + (k - 5) * 8 + j);
}

// ---------------------------------------------------------------------------
// Fused per-slot kernel: features -> layer1 -> WMMA GEMM vs folded weights
// (with per-f feat scaling + bias fused on the D fragment) -> k-reduction ->
// fc1/fc2 head. One workgroup (256 thr = 8 wave32) per (b, agent) slot.
// ---------------------------------------------------------------------------
__global__ void __launch_bounds__(256)
agent_slot_kernel(const float* __restrict__ inputs,
                  const float* __restrict__ actions,
                  const float* __restrict__ w1A, const float* __restrict__ b1A,
                  const float* __restrict__ w1E, const float* __restrict__ b1E,
                  const float* __restrict__ w1L, const float* __restrict__ b1L,
                  const float* __restrict__ fc1_w, const float* __restrict__ fc1_b,
                  const float* __restrict__ fc2_w, const float* __restrict__ fc2_b,
                  const __bf16* __restrict__ W2rbT, const float* __restrict__ b2r,
                  float* __restrict__ out)
{
    __shared__ float sSin[8], sCos[8], sSinA[8], sCosA[8], sVx[8], sVy[8];
    __shared__ float sRbf[8][3];
    __shared__ float sAct[3];                      // act_norm, sinB, cosB
    __shared__ float sFeat[80][10];                // featext (f=9 -> 1.0)
    __shared__ __align__(16) __bf16 sHA[80][64];   // layer-1 activations (A)
    __shared__ float sY[80][64];                   // per-pair merged y
    __shared__ float sM[8][64];                    // m_jki
    __shared__ float sX[8][64];                    // fc1 output
    __shared__ float sQ[8];

    const int tid = threadIdx.x;
    const int s   = blockIdx.x;                    // slot = b*NA + a
    const float* ob = inputs + (size_t)s * 30;

    // warm L2/WGP$ for folded weights (gfx1250 global_prefetch_b8)
    __builtin_prefetch(W2rbT + (size_t)tid * 480, 0, 0);

    // ---- per-entity quantities ----
    if (tid < 8) {
        int e = tid;
        float px, py, vx, vy;
        if (e < 2)      { px = ob[10 + 2 * e]; py = ob[11 + 2 * e];
                          vx = ob[20 + 2 * e]; vy = ob[21 + 2 * e]; }
        else if (e < 5) { int i = e - 2;
                          px = ob[14 + 2 * i]; py = ob[15 + 2 * i];
                          vx = ob[24 + 2 * i]; vy = ob[25 + 2 * i]; }
        else            { int i = e - 5;
                          px = ob[4 + 2 * i];  py = ob[5 + 2 * i];
                          vx = -ob[0];         vy = -ob[1]; }
        float dist = sqrtf(px * px + py * py);
        sSin[e] = py / (dist + EPSF);
        sCos[e] = px / (dist + EPSF);
        float vn = sqrtf(vx * vx + vy * vy);
        sSinA[e] = vy / (vn + EPSF);
        sCosA[e] = vx / (vn + EPSF);
        sVx[e] = vx; sVy[e] = vy;
        sRbf[e][0] = __expf(-0.02f * dist * dist);
        sRbf[e][1] = __expf(-0.02f * (dist - 5.0f) * (dist - 5.0f));
        sRbf[e][2] = __expf(-0.02f * (dist - 10.0f) * (dist - 10.0f));
    } else if (tid == 8) {
        float ax = actions[2 * s], ay = actions[2 * s + 1];
        float an = sqrtf(ax * ax + ay * ay);
        sAct[0] = an;
        sAct[1] = ay / (an + EPSF);
        sAct[2] = ax / (an + EPSF);
    }
    __syncthreads();

    // ---- pair features (featext, f=9 is the constant 1) ----
    for (int idx = tid; idx < 800; idx += 256) {
        int row = idx / 10, f = idx % 10;
        int k, j; float v = 0.0f;
        if (rowKJ(row, k, j)) {
            if      (f <  3) v = sRbf[k][f];
            else if (f == 3) v = sSin[k]  * sCos[j] - sCos[k]  * sSin[j];
            else if (f == 4) v = sCos[k]  * sCos[j] + sSin[k]  * sSin[j];
            else if (f == 5) v = sSinA[k] * sCos[j] - sCosA[k] * sSin[j];
            else if (f == 6) v = sCosA[k] * sCos[j] + sSinA[k] * sSin[j];
            else if (f == 7) v = sVx[k] - sVx[j];
            else if (f == 8) v = sVy[k] - sVy[j];
            else             v = 1.0f;
        }
        sFeat[row][f] = v;
    }
    __syncthreads();

    // ---- layer 1: h = leaky_relu(feat @ W1_net + b1_net), stored as bf16 A ----
    for (int idx = tid; idx < 80 * 64; idx += 256) {
        int row = idx >> 6, m = idx & 63;
        int k, j;
        __bf16 hv = (__bf16)0.0f;
        if (rowKJ(row, k, j)) {
            int net = (row < 16) ? 0 : (row < 48) ? 1 : 2;
            const float* W1 = (net == 0) ? w1A : (net == 1) ? w1E : w1L;
            const float* B1 = (net == 0) ? b1A : (net == 1) ? b1E : b1L;
            float acc = B1[m];
#pragma unroll
            for (int f = 0; f < 9; ++f) acc += sFeat[row][f] * W1[f * 64 + m];
            acc = (acc > 0.0f) ? acc : 0.01f * acc;
            hv = (__bf16)acc;
        }
        sHA[row][m] = hv;
    }
    __syncthreads();

    // ---- WMMA GEMM: per (M-tile, hid-block) combo, loop f=0..9 over folded
    //      weight slices, scale D by featext[row][f] and add folded bias ----
    {
        const int lane   = tid & 31;
        const int wv     = tid >> 5;
        const int nLow   = lane & 15;
        const int laneHi = lane >> 4;     // 0: lanes 0-15, 1: lanes 16-31

        for (int combo = wv; combo < 20; combo += 8) {
            const int mt = combo >> 2;                 // M-tile 0..4
            const int hb = combo & 3;                  // hid block 0..3
            const int net = (mt == 0) ? 0 : (mt < 3) ? 1 : 2;
            const __bf16* Bbase = W2rbT + (size_t)net * 640 * 64;

            v8f acc = {};
#pragma unroll
            for (int f = 0; f < 10; ++f) {
                const int ncol = f * 64 + hb * 16 + nLow;   // N column (f, hid)
                v8f d = {};
#pragma unroll
                for (int ks = 0; ks < 2; ++ks) {
                    // A fragment (16x32 bf16): lanes 0-15 hold M=lane,
                    // K = ks*32 + {0..7, 16..23}; lanes 16-31: K + 8.
                    const int Mrow  = mt * 16 + nLow;
                    const int kbase = ks * 32 + (laneHi ? 8 : 0);
                    v8bf alo = *(const v8bf*)&sHA[Mrow][kbase];
                    v8bf ahi = *(const v8bf*)&sHA[Mrow][kbase + 16];
                    v16bf a;
#pragma unroll
                    for (int i = 0; i < 8; ++i) { a[i] = alo[i]; a[8 + i] = ahi[i]; }

                    // B fragment (32x16 bf16): lanes 0-15 hold K={0..7, 16..23} of
                    // column N=lane, lanes 16-31: K + 8 (N-major store
                    // makes these contiguous 8-element runs).
                    const __bf16* bp = Bbase + (size_t)ncol * 64 + ks * 32 + laneHi * 8;
                    v8bf blo = *(const v8bf*)bp;
                    v8bf bhi = *(const v8bf*)(bp + 16);
                    v16bf b;
#pragma unroll
                    for (int i = 0; i < 8; ++i) { b[i] = blo[i]; b[8 + i] = bhi[i]; }

                    d = __builtin_amdgcn_wmma_f32_16x16x32_bf16(
                            false, a, false, b, (short)0, d, false, false);
                }
                // D layout: lane half selects M base (0/8), vgpr v -> M=v(+8),
                // N = lane&15. Scale by featext and fold reduced bias.
                const float bv = b2r[net * 640 + f * 64 + hb * 16 + nLow];
#pragma unroll
                for (int v = 0; v < 8; ++v) {
                    const int Mrow = mt * 16 + laneHi * 8 + v;
                    const float sc = sFeat[Mrow][f];
                    acc[v] += sc * (d[v] + bv);
                }
            }
#pragma unroll
            for (int v = 0; v < 8; ++v)
                sY[mt * 16 + laneHi * 8 + v][hb * 16 + nLow] = acc[v];
        }
    }
    __syncthreads();

    // ---- reduce over entity axis k: m_jki[j][o] = sum_k y[row(k,j)][o] ----
    for (int idx = tid; idx < 512; idx += 256) {
        int j = idx >> 6, o = idx & 63;
        float sum = 0.0f;
#pragma unroll
        for (int k = 0; k < 8; ++k) sum += sY[rowOf(k, j)][o];
        sM[j][o] = sum;
    }
    __syncthreads();

    // ---- head: x = relu(relu(final) @ fc1 + b), final = [m_jki, an, act_angle]
    for (int idx = tid; idx < 512; idx += 256) {
        int j = idx >> 6, o = idx & 63;
        float acc = fc1_b[o];
        for (int c = 0; c < 64; ++c)
            acc += fmaxf(sM[j][c], 0.0f) * fc1_w[c * 64 + o];
        acc += fmaxf(sAct[0], 0.0f) * fc1_w[64 * 64 + o];
        float aa0 = sAct[1] * sCos[j] - sAct[2] * sSin[j];
        float aa1 = sAct[2] * sCos[j] + sAct[1] * sSin[j];
        acc += fmaxf(aa0, 0.0f) * fc1_w[65 * 64 + o];
        acc += fmaxf(aa1, 0.0f) * fc1_w[66 * 64 + o];
        float xv = fmaxf(acc, 0.0f);
        sX[j][o] = xv;
        out[384 + ((size_t)s * 8 + j) * 64 + o] = xv;
    }
    __syncthreads();

    // ---- q = mean_j (x[j] . fc2_w) + fc2_b ----
    if (tid < 8) {
        float qq = 0.0f;
        for (int o = 0; o < 64; ++o) qq += sX[tid][o] * fc2_w[o];
        sQ[tid] = qq;
    }
    __syncthreads();
    if (tid == 0) {
        float qq = 0.0f;
#pragma unroll
        for (int j = 0; j < 8; ++j) qq += sQ[j];
        out[s] = qq * 0.125f + fc2_b[0];
    }
}

extern "C" void kernel_launch(void* const* d_in, const int* in_sizes, int n_in,
                              void* d_out, int out_size, void* d_ws, size_t ws_size,
                              hipStream_t stream)
{
    (void)in_sizes; (void)n_in; (void)out_size; (void)ws_size;

    const float* inputs   = (const float*)d_in[0];
    // d_in[1] = hidden_state (unused by the reference forward)
    const float* actions  = (const float*)d_in[2];
    const float* hA1_w = (const float*)d_in[3];
    const float* hA1_b = (const float*)d_in[4];
    const float* hA2_w = (const float*)d_in[5];
    const float* hA2_b = (const float*)d_in[6];
    const float* hE1_w = (const float*)d_in[7];
    const float* hE1_b = (const float*)d_in[8];
    const float* hE2_w = (const float*)d_in[9];
    const float* hE2_b = (const float*)d_in[10];
    const float* hL1_w = (const float*)d_in[11];
    const float* hL1_b = (const float*)d_in[12];
    const float* hL2_w = (const float*)d_in[13];
    const float* hL2_b = (const float*)d_in[14];
    const float* merger_w = (const float*)d_in[15];
    const float* fc1_w = (const float*)d_in[16];
    const float* fc1_b = (const float*)d_in[17];
    const float* fc2_w = (const float*)d_in[18];
    const float* fc2_b = (const float*)d_in[19];

    float* out = (float*)d_out;

    __bf16* W2rbT = reinterpret_cast<__bf16*>(d_ws);                     // 3*640*64 bf16
    float*  b2r   = reinterpret_cast<float*>((char*)d_ws + 3 * 640 * 64 * 2); // 3*640 f32

    const int prepThreads = 3 * 650 * 64;
    fold_weights_kernel<<<(prepThreads + 255) / 256, 256, 0, stream>>>(
        merger_w, hA2_w, hA2_b, hE2_w, hE2_b, hL2_w, hL2_b, W2rbT, b2r);

    agent_slot_kernel<<<384, 256, 0, stream>>>(
        inputs, actions,
        hA1_w, hA1_b, hE1_w, hE1_b, hL1_w, hL1_b,
        fc1_w, fc1_b, fc2_w, fc2_b,
        W2rbT, b2r, out);
}
